// fb_SHD_BN_14989435863738
// MI455X (gfx1250) — matrix operand order repaired and checked by reading, not compiled, not run on hardware
//
#include <hip/hip_runtime.h>

// ---------------------------------------------------------------------------
// Types for CDNA5 WMMA (wave32): v_wmma_f32_16x16x32_bf16
// ---------------------------------------------------------------------------
typedef __attribute__((ext_vector_type(16))) __bf16 v16bf;
typedef __attribute__((ext_vector_type(8)))  __bf16 v8bf;
typedef __attribute__((ext_vector_type(4)))  __bf16 v4bf;
typedef __attribute__((ext_vector_type(8)))  float  v8f;
typedef __attribute__((ext_vector_type(4)))  float  v4f;

#define B_SZ   256
#define T_SZ   250
#define IN_SZ  700
#define IN_PAD 704
#define H_SZ   128
#define OUT_SZ 20
#define M_SZ   (B_SZ * T_SZ)   // 64000 rows of the input-projection GEMM
#define NKC    (IN_PAD / 32)   // 22 K-chunks
#define YSTRIDE 136            // LDS row stride (halfs), 272B = multiple of 16

// A-matrix fragment (16x32 bf16, M rows).  Lane L: m=L%16, hl=L/16.
// Element j holds K = 16*(j/8) + hl*8 + (j%8)  ->  two contiguous 16B loads.
__device__ __forceinline__ v16bf load_fragA(const __bf16* base, int stride,
                                            int r0, int kbase, int lane) {
    int m  = lane & 15;
    int hl = lane >> 4;
    const __bf16* p = base + (size_t)(r0 + m) * stride + kbase + hl * 8;
    v8bf lo = *(const v8bf*)p;          // K = hl*8 + 0..7
    v8bf hi = *(const v8bf*)(p + 16);   // K = 16 + hl*8 + 0..7
    return __builtin_shufflevector(lo, hi, 0,1,2,3,4,5,6,7,8,9,10,11,12,13,14,15);
}

// B-matrix fragment (32x16 bf16, K x N).
// Element j holds K = 16*(j/8) + hl*8 + (j%8) (matches fragA).
// Memory is row-major W[n][k] (so B[k][n] = W[n0+n][kbase+k]).
__device__ __forceinline__ v16bf load_fragB(const __bf16* base, int stride,
                                            int n0, int kbase, int lane) {
    int n  = lane & 15;
    int hl = lane >> 4;
    const __bf16* p = base + (size_t)(n0 + n) * stride + kbase + hl * 8;
    v8bf lo = *(const v8bf*)p;          // K = hl*8 + 0..7
    v8bf hi = *(const v8bf*)(p + 16);   // K = 16 + hl*8 + 0..7
    return __builtin_shufflevector(lo, hi, 0,1,2,3,4,5,6,7,8,9,10,11,12,13,14,15);
}

__device__ __forceinline__ v8f wmma_bf16(v16bf a, v16bf b, v8f c) {
    return __builtin_amdgcn_wmma_f32_16x16x32_bf16(
        /*neg_a=*/false, a, /*neg_b=*/false, b,
        /*c_mod=*/(short)0, c, /*reuse_a=*/false, /*reuse_b=*/false);
}

// ---------------------------------------------------------------------------
// Kernel 1: fold BN into weights, cast weights to bf16, build fused biases.
// ---------------------------------------------------------------------------
__global__ __launch_bounds__(128) void prep_kernel(
    const float* __restrict__ w1,  const float* __restrict__ b1,
    const float* __restrict__ g1,  const float* __restrict__ be1,
    const float* __restrict__ rm1, const float* __restrict__ rv1,
    const float* __restrict__ w2,  const float* __restrict__ b2,
    const float* __restrict__ g2,  const float* __restrict__ be2,
    const float* __restrict__ rm2, const float* __restrict__ rv2,
    const float* __restrict__ hw1, const float* __restrict__ hw2,
    const float* __restrict__ hb2,
    __bf16* __restrict__ w1e,  __bf16* __restrict__ hw1b,
    __bf16* __restrict__ w2eb, __bf16* __restrict__ hw2b,
    float* __restrict__ c1,    float* __restrict__ bias2) {
    int h = blockIdx.x;
    float inv1 = g1[h] * rsqrtf(rv1[h] + 1e-5f);
    float inv2 = g2[h] * rsqrtf(rv2[h] + 1e-5f);
    for (int k = threadIdx.x; k < IN_PAD; k += blockDim.x)
        w1e[(size_t)h * IN_PAD + k] =
            (__bf16)((k < IN_SZ) ? w1[(size_t)h * IN_SZ + k] * inv1 : 0.0f);
    for (int k = threadIdx.x; k < H_SZ; k += blockDim.x) {
        hw1b[h * H_SZ + k] = (__bf16)hw1[h * H_SZ + k];
        w2eb[h * H_SZ + k] = (__bf16)(w2[h * H_SZ + k] * inv2);
        hw2b[h * H_SZ + k] = (__bf16)hw2[h * H_SZ + k];
    }
    if (threadIdx.x == 0) {
        c1[h]    = (b1[h] - rm1[h]) * inv1 + be1[h];
        bias2[h] = (b2[h] - rm2[h]) * inv2 + be2[h] + hb2[h];
    }
}

// ---------------------------------------------------------------------------
// Kernel 2: A1[64000,128] = x[64000,700] @ w1e^T + c1   (bf16 WMMA, f32 acc)
// 500 workgroups, 256 threads (8 waves: 2 M-waves x 4 N-waves).
// Software-pipelined: chunk kc+1's global loads (4x b128 NT for x, 4x b64
// for w1e) are issued into registers before chunk kc's WMMAs, so HBM latency
// hides behind matrix + LDS work.  Only the tail chunk (k>=672) carries a
// predicate; all main-loop loads are branch-free.
// ---------------------------------------------------------------------------
__global__ __launch_bounds__(256) void in_proj_kernel(
    const float*  __restrict__ x,    // [M_SZ, 700]
    const __bf16* __restrict__ w1e,  // [128, 704]
    const float*  __restrict__ c1,   // [128]
    float* __restrict__ A1) {        // [M_SZ, 128]
    __shared__ __align__(16) __bf16 xs[128 * 48];
    __shared__ __align__(16) __bf16 wls[128 * 48];
    int tid  = threadIdx.x;
    int lane = tid & 31;
    int wave = tid >> 5;
    int mw   = wave >> 2;   // 0..1 : M rows mw*64 .. mw*64+63
    int nwv  = wave & 3;    // 0..3 : N cols nwv*32 .. nwv*32+31
    size_t row0 = (size_t)blockIdx.x * 128;

    int c4 = (tid & 7) * 4;   // staging column group (4 elems)
    int r0 = tid >> 3;        // staging row 0..31 (4 passes of 32 rows)

    const float*  xbase = x + (row0 + r0) * IN_SZ + c4;   // + p*32*IN_SZ + k0
    const __bf16* wbase = w1e + (size_t)r0 * IN_PAD + c4; // + p*32*IN_PAD + k0
    bool tail_ok = (c4 < 28);   // tail chunk: k = 672+c4 .. 672+c4+3 valid?

    v8f acc[4][2];
#pragma unroll
    for (int ms = 0; ms < 4; ++ms)
#pragma unroll
        for (int ns = 0; ns < 2; ++ns) acc[ms][ns] = {};

    // ---- prologue: prefetch chunk 0 into registers (branch-free) ----
    v4f  xv[4];
    v4bf wv[4];
#pragma unroll
    for (int p = 0; p < 4; ++p) {
        xv[p] = __builtin_nontemporal_load((const v4f*)(xbase + (size_t)p * 32 * IN_SZ));
        wv[p] = *(const v4bf*)(wbase + (size_t)p * 32 * IN_PAD);
    }

#pragma unroll 1
    for (int kc = 0; kc < NKC; ++kc) {
        // ---- commit staged registers to LDS ----
#pragma unroll
        for (int p = 0; p < 4; ++p) {
            v4bf b;
            b[0] = (__bf16)xv[p][0]; b[1] = (__bf16)xv[p][1];
            b[2] = (__bf16)xv[p][2]; b[3] = (__bf16)xv[p][3];
            *(v4bf*)&xs[(r0 + p * 32) * 48 + c4] = b;
            *(v4bf*)&wls[(r0 + p * 32) * 48 + c4] = wv[p];
        }
        __syncthreads();

        // ---- prefetch chunk kc+1 (overlaps the WMMAs below) ----
        if (kc + 1 < NKC) {
            int k1 = (kc + 1) * 32;
            if (kc + 1 < NKC - 1) {           // fully in-bounds: branch-free
#pragma unroll
                for (int p = 0; p < 4; ++p)
                    xv[p] = __builtin_nontemporal_load(
                        (const v4f*)(xbase + (size_t)p * 32 * IN_SZ + k1));
            } else {                          // tail chunk: predicated once
#pragma unroll
                for (int p = 0; p < 4; ++p) {
                    v4f v = {0.0f, 0.0f, 0.0f, 0.0f};
                    if (tail_ok)
                        v = __builtin_nontemporal_load(
                            (const v4f*)(xbase + (size_t)p * 32 * IN_SZ + k1));
                    xv[p] = v;
                }
            }
#pragma unroll
            for (int p = 0; p < 4; ++p)       // w1e pre-padded: branch-free
                wv[p] = *(const v4bf*)(wbase + (size_t)p * 32 * IN_PAD + k1);
        }

        // ---- fragments + 8 WMMAs for chunk kc ----
        v16bf bfr[2];
#pragma unroll
        for (int ns = 0; ns < 2; ++ns)
            bfr[ns] = load_fragB(wls, 48, nwv * 32 + ns * 16, 0, lane);
#pragma unroll
        for (int ms = 0; ms < 4; ++ms) {
            v16bf afr = load_fragA(xs, 48, (mw * 4 + ms) * 16, 0, lane);
#pragma unroll
            for (int ns = 0; ns < 2; ++ns)
                acc[ms][ns] = wmma_bf16(afr, bfr[ns], acc[ms][ns]);
        }
        __syncthreads();
    }

    // store D tiles: element r of lane L -> (m = r + 8*(L/16), n = L%16)
    int n  = lane & 15;
    int hl = lane >> 4;
#pragma unroll
    for (int ms = 0; ms < 4; ++ms)
#pragma unroll
        for (int ns = 0; ns < 2; ++ns) {
            int col = nwv * 32 + ns * 16 + n;
            float cc = c1[col];
#pragma unroll
            for (int r = 0; r < 8; ++r) {
                int m = (mw * 4 + ms) * 16 + r + 8 * hl;
                A1[(row0 + m) * H_SZ + col] = acc[ms][ns][r] + cc;
            }
        }
}

// ---------------------------------------------------------------------------
// Kernel 3: the sequential LIF scan.  16 workgroups (16 batch rows each),
// 8 waves; wave w owns hidden columns [16w, 16w+16).  Spikes ping-pong in
// LDS (bf16); recurrent weights live in VGPRs as 12 resident B-fragments.
// Per step per wave: 12 x v_wmma_f32_16x16x32_bf16 + VALU LIF.
// ---------------------------------------------------------------------------
__global__ __launch_bounds__(256) void snn_scan_kernel(
    const float*  __restrict__ A1,    // [B,T,H] fused input projection
    const __bf16* __restrict__ hw1b,  // [H,H]
    const __bf16* __restrict__ w2eb,  // [H,H] (BN folded)
    const __bf16* __restrict__ hw2b,  // [H,H]
    const float*  __restrict__ hb1,   // [H]
    const float*  __restrict__ bias2, // [H] fused layer-2 bias
    float* __restrict__ ysum2) {      // [B,H] spike counts
    __shared__ __align__(16) __bf16 y1buf[2][16 * YSTRIDE];
    __shared__ __align__(16) __bf16 y2buf[2][16 * YSTRIDE];

    int tid  = threadIdx.x;
    int lane = tid & 31;
    int wave = tid >> 5;
    int h0   = wave * 16;
    int n    = lane & 15;
    int hl   = lane >> 4;
    int b0   = blockIdx.x * 16;

    // resident weight fragments (K = 128 -> 4 frags of K32 per matrix)
    v16bf Bh1[4], Bw2[4], Bh2[4];
#pragma unroll
    for (int f = 0; f < 4; ++f) {
        Bh1[f] = load_fragB(hw1b, H_SZ, h0, f * 32, lane);
        Bw2[f] = load_fragB(w2eb, H_SZ, h0, f * 32, lane);
        Bh2[f] = load_fragB(hw2b, H_SZ, h0, f * 32, lane);
    }
    float bias1v = hb1[h0 + n];
    float bias2v = bias2[h0 + n];

    for (int i = tid; i < 16 * YSTRIDE; i += 256) {
        y1buf[0][i] = (__bf16)0.0f; y1buf[1][i] = (__bf16)0.0f;
        y2buf[0][i] = (__bf16)0.0f; y2buf[1][i] = (__bf16)0.0f;
    }
    v8f u1 = {}, u2 = {}, ys = {};
    __syncthreads();

    int cur = 0;
    for (int t = 0; t < T_SZ; ++t) {
        int nxt = cur ^ 1;

        // ---- issue this step's a1 loads first (independent of LDS/WMMA) ----
        float a1v[8];
#pragma unroll
        for (int r = 0; r < 8; ++r) {
            int m = r + 8 * hl;
            a1v[r] = A1[(((size_t)(b0 + m)) * T_SZ + t) * H_SZ + h0 + n];
        }

        // ---- layer 1: i1 = y1_prev @ hw1^T + a1 + hb1 ----
        v8f acc1 = {};
#pragma unroll
        for (int f = 0; f < 4; ++f) {
            v16bf a = load_fragA(&y1buf[cur][0], YSTRIDE, 0, f * 32, lane);
            acc1 = wmma_bf16(a, Bh1[f], acc1);
        }
#pragma unroll
        for (int r = 0; r < 8; ++r) {
            int m = r + 8 * hl;
            float i1 = acc1[r] + a1v[r] + bias1v;
            float u  = u1[r] + (i1 - u1[r]) * 0.5f;   // TAU = 2
            float y  = (u >= 1.0f) ? 1.0f : 0.0f;     // VTH = 1
            u1[r] = u * (1.0f - y);                   // hard reset
            y1buf[nxt][m * YSTRIDE + h0 + n] = (__bf16)y;
        }
        __syncthreads();

        // ---- layer 2: i2 = y1_new @ w2e^T + y2_prev @ hw2^T + bias2 ----
        v8f acc2 = {};
#pragma unroll
        for (int f = 0; f < 4; ++f) {
            v16bf a = load_fragA(&y1buf[nxt][0], YSTRIDE, 0, f * 32, lane);
            acc2 = wmma_bf16(a, Bw2[f], acc2);
        }
#pragma unroll
        for (int f = 0; f < 4; ++f) {
            v16bf a = load_fragA(&y2buf[cur][0], YSTRIDE, 0, f * 32, lane);
            acc2 = wmma_bf16(a, Bh2[f], acc2);
        }
#pragma unroll
        for (int r = 0; r < 8; ++r) {
            int m = r + 8 * hl;
            float i2 = acc2[r] + bias2v;
            float u  = u2[r] + (i2 - u2[r]) * 0.5f;
            float y  = (u >= 1.0f) ? 1.0f : 0.0f;
            u2[r] = u * (1.0f - y);
            ys[r] += y;
            y2buf[nxt][m * YSTRIDE + h0 + n] = (__bf16)y;
        }
        __syncthreads();
        cur = nxt;
    }

#pragma unroll
    for (int r = 0; r < 8; ++r) {
        int m = r + 8 * hl;
        ysum2[(size_t)(b0 + m) * H_SZ + h0 + n] = ys[r];
    }
}

// ---------------------------------------------------------------------------
// Kernel 4: out[b,o] = ysum2[b,:] . wo[o,:] + T * bo[o]
// ---------------------------------------------------------------------------
__global__ __launch_bounds__(256) void out_proj_kernel(
    const float* __restrict__ ysum2, const float* __restrict__ wo,
    const float* __restrict__ bo, float* __restrict__ out) {
    int idx = blockIdx.x * blockDim.x + threadIdx.x;
    if (idx >= B_SZ * OUT_SZ) return;
    int b = idx / OUT_SZ, o = idx % OUT_SZ;
    float s = 0.0f;
#pragma unroll 8
    for (int h = 0; h < H_SZ; ++h)
        s += ysum2[b * H_SZ + h] * wo[o * H_SZ + h];
    out[idx] = s + (float)T_SZ * bo[o];
}

// ---------------------------------------------------------------------------
extern "C" void kernel_launch(void* const* d_in, const int* in_sizes, int n_in,
                              void* d_out, int out_size, void* d_ws, size_t ws_size,
                              hipStream_t stream) {
    const float* x   = (const float*)d_in[0];
    const float* w1  = (const float*)d_in[1];
    const float* b1  = (const float*)d_in[2];
    const float* g1  = (const float*)d_in[3];
    const float* be1 = (const float*)d_in[4];
    const float* rm1 = (const float*)d_in[5];
    const float* rv1 = (const float*)d_in[6];
    const float* hw1 = (const float*)d_in[7];
    const float* hb1 = (const float*)d_in[8];
    const float* w2  = (const float*)d_in[9];
    const float* b2  = (const float*)d_in[10];
    const float* g2  = (const float*)d_in[11];
    const float* be2 = (const float*)d_in[12];
    const float* rm2 = (const float*)d_in[13];
    const float* rv2 = (const float*)d_in[14];
    const float* hw2 = (const float*)d_in[15];
    const float* hb2 = (const float*)d_in[16];
    const float* wo  = (const float*)d_in[17];
    const float* bo  = (const float*)d_in[18];
    float* out = (float*)d_out;

    // workspace carve-out (all offsets 256B aligned)
    char* p = (char*)d_ws;
    float*  A1    = (float*)p;  p += (size_t)M_SZ * H_SZ * sizeof(float);   // 32.77 MB
    __bf16* w1e   = (__bf16*)p; p += (size_t)H_SZ * IN_PAD * 2;             // 176 KB
    __bf16* hw1b  = (__bf16*)p; p += (size_t)H_SZ * H_SZ * 2;
    __bf16* w2eb  = (__bf16*)p; p += (size_t)H_SZ * H_SZ * 2;
    __bf16* hw2b  = (__bf16*)p; p += (size_t)H_SZ * H_SZ * 2;
    float*  c1    = (float*)p;  p += 512;
    float*  bias2 = (float*)p;  p += 512;
    float*  ysum2 = (float*)p;  p += (size_t)B_SZ * H_SZ * sizeof(float);

    prep_kernel<<<H_SZ, 128, 0, stream>>>(w1, b1, g1, be1, rm1, rv1,
                                          w2, b2, g2, be2, rm2, rv2,
                                          hw1, hw2, hb2,
                                          w1e, hw1b, w2eb, hw2b, c1, bias2);
    in_proj_kernel<<<M_SZ / 128, 256, 0, stream>>>(x, w1e, c1, A1);
    snn_scan_kernel<<<B_SZ / 16, 256, 0, stream>>>(A1, hw1b, w2eb, hw2b,
                                                   hb1, bias2, ysum2);
    out_proj_kernel<<<(B_SZ * OUT_SZ + 255) / 256, 256, 0, stream>>>(ysum2, wo, bo, out);
}
